// PELA_6270652252759
// MI455X (gfx1250) — compile-verified, matrix operand order checked by reading, not run on hardware
//
#include <hip/hip_runtime.h>
#include <math.h>

// ---------------------------------------------------------------------------
// PELA focused-linear-attention block for MI455X (gfx1250, wave32, WMMA).
// All heavy GEMMs use v_wmma_f32_16x16x32_f16 (fp32 accumulation).
// Projection GEMM stages its tiles with the Tensor Data Mover (TDM).
// ---------------------------------------------------------------------------

typedef __attribute__((ext_vector_type(16))) _Float16 v16h;
typedef __attribute__((ext_vector_type(8)))  _Float16 v8h;
typedef __attribute__((ext_vector_type(8)))  float    v8f;

#define BDIM   8
#define C_DIM  512
#define NPIX   4096      // 64*64
#define NH     8
#define HD     64
#define HDIM_  64
#define WDIM_  64

// LDS tile row stride: 40 halfs = 80 bytes (16B aligned rows -> b128 LDS ops,
// and matches the TDM pad_interval/pad_amount encoding below).
#define LPAD 40

// Fragment index helper (ISA 7.12.2, 16-bit A 16x32 layout):
// element e (0..15) of the v16h fragment maps to K = (e<8 ? e : e+8) + 8*(lane>>4)
// -> per lane two contiguous 16B runs; backend vectorizes to ds_load_2addr_b64.
__device__ __forceinline__ int frag_k(int e, int grp) {
    return (e < 8 ? e : e + 8) + (grp << 3);
}

#if defined(__HIP_DEVICE_COMPILE__) && __has_builtin(__builtin_amdgcn_tensor_load_to_lds)
#define USE_TDM 1
#else
#define USE_TDM 0
#endif

// Issue one 2-D TDM tile load: tile (tile_d1 rows x tile_d0 elems of 2B) from a
// row-major f16 tensor with row stride stride0 (elems), landing in LDS with a
// 16-byte pad after every 64 data bytes (i.e. LDS row stride = 80B = LPAD halfs).
// D# layout per CDNA5 ISA ch.8 (group0 128b, group1 256b; groups 2/3 zero => 2D).
__device__ __forceinline__ void tdm_load_2d(unsigned lds_off, const void* gptr,
                                            unsigned stride0,
                                            unsigned tile_d0, unsigned tile_d1) {
#if USE_TDM
    typedef unsigned u32x4 __attribute__((ext_vector_type(4)));
    typedef int i32x8 __attribute__((ext_vector_type(8)));
    typedef int i32x4 __attribute__((ext_vector_type(4)));
    unsigned long long ga = (unsigned long long)gptr;
    u32x4 g0;
    g0[0] = 1u;                                   // count=1, user mode
    g0[1] = lds_off;                              // lds_addr (bytes)
    g0[2] = (unsigned)ga;                         // global_addr[31:0]
    g0[3] = (unsigned)(ga >> 32) | 0x80000000u;   // global_addr[56:32] | type=2
    // group1: data_size=1 (2B) <<16 | pad_enable(b20) | pad_interval=3 (16 DW)
    //         | pad_amount=3 (4 DW)
    unsigned w0 = 0x10000u | (1u << 20) | (3u << 22) | (3u << 25);
    unsigned td0 = tile_d0, td1 = tile_d1;        // tensor dims = tile dims
    i32x8 g1;
    g1[0] = (int)w0;
    g1[1] = (int)(td0 << 16);                     // tensor_dim0[15:0] @bits63:48
    g1[2] = (int)((td0 >> 16) | (td1 << 16));     // tensor_dim0[31:16] | tensor_dim1[15:0]
    g1[3] = (int)((td1 >> 16) | (tile_d0 << 16)); // tensor_dim1[31:16] | tile_dim0
    g1[4] = (int)tile_d1;                         // tile_dim1 (tile_dim2 = 0)
    g1[5] = (int)stride0;                         // tensor_dim0_stride[31:0]
    g1[6] = 0;
    g1[7] = 0;
    i32x4 z4 = {0, 0, 0, 0};
#if __clang_major__ >= 23
    i32x8 z8 = {0, 0, 0, 0, 0, 0, 0, 0};
    __builtin_amdgcn_tensor_load_to_lds(g0, g1, z4, z4, z8, 0);
#else
    __builtin_amdgcn_tensor_load_to_lds(g0, g1, z4, z4, 0);
#endif
#else
    (void)lds_off; (void)gptr; (void)stride0; (void)tile_d0; (void)tile_d1;
#endif
}

// ===========================================================================
// K0: weight conversion + scale + zero init (grid-stride)
// ===========================================================================
__global__ void k_prep(const float* __restrict__ Wq, const float* __restrict__ Wkv,
                       const float* __restrict__ Wproj, const float* __restrict__ scale_param,
                       _Float16* __restrict__ WqkvH, _Float16* __restrict__ WprojH,
                       float* __restrict__ inv_scale, float* __restrict__ ksum,
                       float* __restrict__ kvm) {
    const int NQKV = 1536 * 512;
    const int NPRJ = 512 * 512;
    const int NKVM = NH * BDIM * HD * HD;
    const int total = NQKV + NPRJ + C_DIM + BDIM * C_DIM + NKVM;
    for (int i = blockIdx.x * blockDim.x + threadIdx.x; i < total;
         i += gridDim.x * blockDim.x) {
        int j = i;
        if (j < NQKV) { WqkvH[j] = (_Float16)(j < NPRJ ? Wq[j] : Wkv[j - NPRJ]); continue; }
        j -= NQKV;
        if (j < NPRJ) { WprojH[j] = (_Float16)Wproj[j]; continue; }
        j -= NPRJ;
        if (j < C_DIM) {
            float s = scale_param[j];
            float sp = (s > 20.f) ? s : log1pf(expf(s));   // softplus
            inv_scale[j] = 1.f / sp;
            continue;
        }
        j -= C_DIM;
        if (j < BDIM * C_DIM) { ksum[j] = 0.f; continue; }
        j -= BDIM * C_DIM;
        kvm[j] = 0.f;
    }
}

// ===========================================================================
// K1: fused QKV GEMM.  qkv[b,n,co] = sum_ci x[b,ci,n] * Wqkv[co,ci]
// Tile: 128(M=n) x 64(Nout=co), BK=32.  8 waves: 4 along M x 2 along N.
// q,k written (B,N,C) f32; v written (B,C,N) f32 (conv/kvm friendly).
// ===========================================================================
__global__ __launch_bounds__(256) void k_qkv_gemm(
    const float* __restrict__ x, const _Float16* __restrict__ Wqkv,
    float* __restrict__ q, float* __restrict__ k, float* __restrict__ v) {
    int bid = blockIdx.x;
    int ct = bid % 24; bid /= 24;
    int mt = bid % 32; bid /= 32;
    int b  = bid;
    int n0 = mt * 128, c0 = ct * 64;

    __shared__ _Float16 As[128][LPAD];
    __shared__ _Float16 Bs[64][LPAD];

    int tid = threadIdx.x, lane = tid & 31, w = tid >> 5;
    int wm = w & 3, wn = w >> 2;
    int mrow = lane & 15, grp = lane >> 4;

    v8f acc[2][2] = {};
    const float* xb = x + (size_t)b * C_DIM * NPIX;

    for (int k0 = 0; k0 < C_DIM; k0 += 32) {
        // ---- A tile: x[ci=k0+kk][n0+m], f32 -> f16, transpose-staged.
        // Phase 1: 4x float4 loads per thread (all issued before any use).
        float4 ra[4];
        #pragma unroll
        for (int j = 0; j < 4; ++j) {
            int u = tid + j * 256;               // 0..1023 float4 units
            int kk = u >> 5, mv = (u & 31) * 4;
            ra[j] = *(const float4*)&xb[(size_t)(k0 + kk) * NPIX + n0 + mv];
        }
        if (k0 + 32 < C_DIM)
            __builtin_prefetch(&xb[(size_t)(k0 + 32) * NPIX + n0 + tid], 0, 0);
        // ---- B tile: Wqkv f16, one 16B copy per thread.
        {
            int nc = tid >> 2, kv = (tid & 3) * 8;
            *(v8h*)&Bs[nc][kv] =
                *(const v8h*)&Wqkv[(size_t)(c0 + nc) * C_DIM + k0 + kv];
        }
        // Phase 2: convert + scatter A into LDS.
        #pragma unroll
        for (int j = 0; j < 4; ++j) {
            int u = tid + j * 256;
            int kk = u >> 5, mv = (u & 31) * 4;
            As[mv + 0][kk] = (_Float16)ra[j].x;
            As[mv + 1][kk] = (_Float16)ra[j].y;
            As[mv + 2][kk] = (_Float16)ra[j].z;
            As[mv + 3][kk] = (_Float16)ra[j].w;
        }
        __syncthreads();

        v16h af[2], bf[2];
        #pragma unroll
        for (int t = 0; t < 2; ++t) {
            int mb = wm * 32 + t * 16 + mrow;
            int nb = wn * 32 + t * 16 + mrow;
            #pragma unroll
            for (int e = 0; e < 16; ++e) {
                int kk = frag_k(e, grp);
                af[t][e] = As[mb][kk];
                bf[t][e] = Bs[nb][kk];
            }
        }
        #pragma unroll
        for (int tm = 0; tm < 2; ++tm)
            #pragma unroll
            for (int tn = 0; tn < 2; ++tn)
                acc[tm][tn] = __builtin_amdgcn_wmma_f32_16x16x32_f16(
                    false, af[tm], false, bf[tn], (short)0, acc[tm][tn], false, false);
        __syncthreads();
    }

    #pragma unroll
    for (int tm = 0; tm < 2; ++tm)
        #pragma unroll
        for (int tn = 0; tn < 2; ++tn)
            #pragma unroll
            for (int r = 0; r < 8; ++r) {
                int n  = n0 + wm * 32 + tm * 16 + r + 8 * grp;
                int co = c0 + wn * 32 + tn * 16 + (lane & 15);
                float val = acc[tm][tn][r];
                if (co < 512)
                    q[((size_t)b * NPIX + n) * C_DIM + co] = val;
                else if (co < 1024)
                    k[((size_t)b * NPIX + n) * C_DIM + (co - 512)] = val;
                else
                    v[((size_t)b * C_DIM + (co - 1024)) * NPIX + n] = val;
            }
}

// ===========================================================================
// K2: focused transform: t = (relu(x)+1e-6)/scale; out = t^3 * ||t||/||t^3||
// ===========================================================================
__global__ __launch_bounds__(256) void k_transform(
    const float* __restrict__ src, const float* __restrict__ inv_scale,
    _Float16* __restrict__ dst, float* __restrict__ ksum, int is_k) {
    __shared__ float ssum[C_DIM];
    int bid = blockIdx.x;
    int b = bid >> 7, rt = bid & 127;
    int tid = threadIdx.x, lane = tid & 31, w = tid >> 5;

    if (is_k) {
        for (int i = tid; i < C_DIM; i += 256) ssum[i] = 0.f;
        __syncthreads();
    }

    for (int rr = 0; rr < 4; ++rr) {
        int n = rt * 32 + w * 4 + rr;
        const float* row = src + ((size_t)b * NPIX + n) * C_DIM;
        float t[16], s2 = 0.f, s6 = 0.f;
        #pragma unroll
        for (int i = 0; i < 16; ++i) {
            int c = lane + i * 32;
            float vv = row[c];
            vv = (vv > 0.f ? vv : 0.f) + 1e-6f;
            vv *= inv_scale[c];
            t[i] = vv;
            float p = vv * vv * vv;
            s2 += vv * vv;
            s6 += p * p;
        }
        #pragma unroll
        for (int off = 16; off; off >>= 1) {
            s2 += __shfl_xor(s2, off, 32);
            s6 += __shfl_xor(s6, off, 32);
        }
        float fac = sqrtf(s2) * rsqrtf(s6);   // ||t|| / ||t^3||
        #pragma unroll
        for (int i = 0; i < 16; ++i) {
            int c = lane + i * 32;
            float p = t[i] * t[i] * t[i] * fac;
            if (is_k) {
                dst[((size_t)b * C_DIM + c) * NPIX + n] = (_Float16)p;
                atomicAdd(&ssum[c], p);
            } else {
                dst[((size_t)b * NPIX + n) * C_DIM + c] = (_Float16)p;
            }
        }
    }
    if (is_k) {
        __syncthreads();
        for (int i = tid; i < C_DIM; i += 256)
            atomicAdd(&ksum[b * C_DIM + i], ssum[i]);
    }
}

// ===========================================================================
// K3: z[b,n,h] = 1 / (q' . k_mean + 1e-6)
// ===========================================================================
__global__ void k_zfac(const _Float16* __restrict__ qp, const float* __restrict__ ksum,
                       float* __restrict__ z) {
    int idx = blockIdx.x * blockDim.x + threadIdx.x;
    if (idx >= BDIM * NPIX * NH) return;
    int h = idx & 7, n = (idx >> 3) & (NPIX - 1), b = idx >> 15;
    const _Float16* qr = qp + ((size_t)b * NPIX + n) * C_DIM + h * HD;
    const float* km = ksum + b * C_DIM + h * HD;
    float s = 0.f;
    #pragma unroll 8
    for (int i = 0; i < HD; ++i) s += (float)qr[i] * km[i];
    s *= (1.f / NPIX);
    z[idx] = 1.f / (s + 1e-6f);
}

// ===========================================================================
// K4: kvm[b,h,d',e] += sum_n k'[b,h,n,d'] v[b,h,n,e]   (split-K over n)
// ===========================================================================
__global__ __launch_bounds__(256) void k_kvm_gemm(
    const _Float16* __restrict__ kp, const float* __restrict__ v,
    float* __restrict__ kvm) {
    int bid = blockIdx.x;
    int chunk = bid & 3; bid >>= 2;
    int h = bid & 7;     bid >>= 3;
    int b = bid;
    int nchunk0 = chunk * 1024;

    __shared__ _Float16 As[64][LPAD];
    __shared__ _Float16 Bs[64][LPAD];

    int tid = threadIdx.x, lane = tid & 31, w = tid >> 5;
    int wm = w & 3, wn = w >> 2;
    int mrow = lane & 15, grp = lane >> 4;

    v8f acc[2] = {};
    const _Float16* kb = kp + ((size_t)b * C_DIM + h * HD) * NPIX;
    const float*    vb = v  + ((size_t)b * C_DIM + h * HD) * NPIX;

    int m = tid >> 2, kv = (tid & 3) * 8;
    for (int kk0 = 0; kk0 < 1024; kk0 += 32) {
        int nbase = nchunk0 + kk0;
        // A: f16 16B copy; B: 8 f32 -> 8 f16, 16B store.
        v8h ka = *(const v8h*)&kb[(size_t)m * NPIX + nbase + kv];
        float4 r0 = *(const float4*)&vb[(size_t)m * NPIX + nbase + kv];
        float4 r1 = *(const float4*)&vb[(size_t)m * NPIX + nbase + kv + 4];
        v8h hv;
        hv[0] = (_Float16)r0.x; hv[1] = (_Float16)r0.y;
        hv[2] = (_Float16)r0.z; hv[3] = (_Float16)r0.w;
        hv[4] = (_Float16)r1.x; hv[5] = (_Float16)r1.y;
        hv[6] = (_Float16)r1.z; hv[7] = (_Float16)r1.w;
        *(v8h*)&As[m][kv] = ka;
        *(v8h*)&Bs[m][kv] = hv;
        __syncthreads();

        v16h af;
        int mb = wm * 16 + mrow;
        #pragma unroll
        for (int e = 0; e < 16; ++e) af[e] = As[mb][frag_k(e, grp)];
        #pragma unroll
        for (int tn = 0; tn < 2; ++tn) {
            v16h bf;
            int nb = wn * 32 + tn * 16 + mrow;
            #pragma unroll
            for (int e = 0; e < 16; ++e) bf[e] = Bs[nb][frag_k(e, grp)];
            acc[tn] = __builtin_amdgcn_wmma_f32_16x16x32_f16(
                false, af, false, bf, (short)0, acc[tn], false, false);
        }
        __syncthreads();
    }

    float* dst = kvm + ((size_t)(b * NH + h)) * HD * HD;
    #pragma unroll
    for (int tn = 0; tn < 2; ++tn)
        #pragma unroll
        for (int r = 0; r < 8; ++r) {
            int mr  = wm * 16 + r + 8 * grp;
            int col = wn * 32 + tn * 16 + (lane & 15);
            atomicAdd(&dst[mr * HD + col], acc[tn][r]);
        }
}

__global__ void k_kvm_cvt(const float* __restrict__ kvm, _Float16* __restrict__ kvmH) {
    int i = blockIdx.x * blockDim.x + threadIdx.x;
    if (i < BDIM * NH * HD * HD) kvmH[i] = (_Float16)(kvm[i] * (1.0f / NPIX));
}

// ===========================================================================
// K5: out[b,n,h*64+e] = (q' . kvm) * z   — thin GEMM, K=64, per (b,h).
// ===========================================================================
__global__ __launch_bounds__(256) void k_attn_out(
    const _Float16* __restrict__ qp, const _Float16* __restrict__ kvmH,
    const float* __restrict__ z, float* __restrict__ xat) {
    int bid = blockIdx.x;
    int mt = bid & 31; bid >>= 5;
    int h  = bid & 7;  bid >>= 3;
    int b  = bid;
    int n0 = mt * 128;

    __shared__ _Float16 As[128][LPAD];
    __shared__ _Float16 Bs[64][LPAD];

    int tid = threadIdx.x, lane = tid & 31, w = tid >> 5;
    int wm = w & 3, wn = w >> 2;
    int mrow = lane & 15, grp = lane >> 4;

    v8f acc[2][2] = {};
    const _Float16* qb  = qp + (size_t)b * NPIX * C_DIM + h * HD;
    const _Float16* kvb = kvmH + ((size_t)(b * NH + h)) * HD * HD;

    for (int k0 = 0; k0 < HD; k0 += 32) {
        // A: two 16B copies per thread.
        #pragma unroll
        for (int j = 0; j < 2; ++j) {
            int u = tid + j * 256;
            int m = u >> 2, kv = (u & 3) * 8;
            *(v8h*)&As[m][kv] =
                *(const v8h*)&qb[(size_t)(n0 + m) * C_DIM + k0 + kv];
        }
        // B: transpose-staged scalar copy (coalesced reads along e).
        _Float16 rb[8];
        #pragma unroll
        for (int i = 0; i < 8; ++i) {
            int idx = tid + i * 256;
            int e = idx & 63, kk = idx >> 6;
            rb[i] = kvb[(k0 + kk) * HD + e];
        }
        #pragma unroll
        for (int i = 0; i < 8; ++i) {
            int idx = tid + i * 256;
            int e = idx & 63, kk = idx >> 6;
            Bs[e][kk] = rb[i];
        }
        __syncthreads();

        v16h af[2], bf[2];
        #pragma unroll
        for (int t = 0; t < 2; ++t) {
            int mb = wm * 32 + t * 16 + mrow;
            int nb = wn * 32 + t * 16 + mrow;
            #pragma unroll
            for (int e = 0; e < 16; ++e) {
                int kk = frag_k(e, grp);
                af[t][e] = As[mb][kk];
                bf[t][e] = Bs[nb][kk];
            }
        }
        #pragma unroll
        for (int tm = 0; tm < 2; ++tm)
            #pragma unroll
            for (int tn = 0; tn < 2; ++tn)
                acc[tm][tn] = __builtin_amdgcn_wmma_f32_16x16x32_f16(
                    false, af[tm], false, bf[tn], (short)0, acc[tm][tn], false, false);
        __syncthreads();
    }

    #pragma unroll
    for (int tm = 0; tm < 2; ++tm)
        #pragma unroll
        for (int tn = 0; tn < 2; ++tn)
            #pragma unroll
            for (int r = 0; r < 8; ++r) {
                int n   = n0 + wm * 32 + tm * 16 + r + 8 * grp;
                int col = wn * 32 + tn * 16 + (lane & 15);
                float zv = z[((size_t)b * NPIX + n) * NH + h];
                xat[((size_t)b * NPIX + n) * C_DIM + h * HD + col] = acc[tm][tn][r] * zv;
            }
}

// ===========================================================================
// K_dwc: depthwise 5x5 conv on v (B,C,64,64), weight channel = c % 64.
// ===========================================================================
__global__ __launch_bounds__(256) void k_dwc(
    const float* __restrict__ v, const float* __restrict__ wgt,
    const float* __restrict__ bias, float* __restrict__ out) {
    int bid = blockIdx.x;
    int tile = bid & 15; int bc = bid >> 4;
    int c = bc & (C_DIM - 1);
    int dd = c & (HD - 1);
    int ty0 = (tile >> 2) * 16, tx0 = (tile & 3) * 16;

    __shared__ float sm[20][20];
    __shared__ float sw[25];
    int tid = threadIdx.x;
    if (tid < 25) sw[tid] = wgt[dd * 25 + tid];

    const float* plane = v + (size_t)bc * NPIX;
    float stage[2];
    #pragma unroll
    for (int j = 0; j < 2; ++j) {
        int i = tid + j * 256;
        float val = 0.f;
        if (i < 400) {
            int yy = i / 20 + ty0 - 2, xx = i % 20 + tx0 - 2;
            if (yy >= 0 && yy < HDIM_ && xx >= 0 && xx < WDIM_)
                val = plane[yy * WDIM_ + xx];
        }
        stage[j] = val;
    }
    #pragma unroll
    for (int j = 0; j < 2; ++j) {
        int i = tid + j * 256;
        if (i < 400) sm[i / 20][i % 20] = stage[j];
    }
    __syncthreads();

    int py = tid >> 4, px = tid & 15;
    float s = bias[dd];
    #pragma unroll
    for (int ky = 0; ky < 5; ++ky)
        #pragma unroll
        for (int kx = 0; kx < 5; ++kx)
            s += sm[py + ky][px + kx] * sw[ky * 5 + kx];
    out[(size_t)bc * NPIX + (ty0 + py) * WDIM_ + tx0 + px] = s;
}

// ===========================================================================
// K_add: attn_plus f16(B,N,C) = x_attn(B,N,C) + dwc(B,C,N)  (LDS transpose)
// ===========================================================================
__global__ __launch_bounds__(256) void k_add(
    const float* __restrict__ xat, const float* __restrict__ dwc,
    _Float16* __restrict__ ap) {
    int bid = blockIdx.x;
    int ct = bid & 15; bid >>= 4;
    int nt = bid & 127; bid >>= 7;
    int b = bid;
    __shared__ float sd[32][33];
    int tid = threadIdx.x;
    float stage[4];
    #pragma unroll
    for (int j = 0; j < 4; ++j) {
        int i = tid + j * 256;
        int cc = i >> 5, nn = i & 31;
        stage[j] = dwc[((size_t)b * C_DIM + ct * 32 + cc) * NPIX + nt * 32 + nn];
    }
    #pragma unroll
    for (int j = 0; j < 4; ++j) {
        int i = tid + j * 256;
        sd[i >> 5][i & 31] = stage[j];
    }
    __syncthreads();
    #pragma unroll
    for (int j = 0; j < 4; ++j) {
        int i = tid + j * 256;
        int nn = i >> 5, cc = i & 31;
        size_t o = ((size_t)b * NPIX + nt * 32 + nn) * C_DIM + ct * 32 + cc;
        ap[o] = (_Float16)(xat[o] + sd[cc][nn]);
    }
}

// ===========================================================================
// K6: projection GEMM + bias, output transposed to (B,C,N).
// Tiles staged by the Tensor Data Mover (one wave issues two D# loads,
// waits TENSORcnt, then the block barrier publishes the LDS tiles).
// ===========================================================================
__global__ __launch_bounds__(256) void k_proj_gemm(
    const _Float16* __restrict__ ap, const _Float16* __restrict__ Wp,
    const float* __restrict__ bproj, float* __restrict__ out) {
    int bid = blockIdx.x;
    int ct = bid & 7;  bid >>= 3;
    int mt = bid & 31; bid >>= 5;
    int b = bid;
    int n0 = mt * 128, c0 = ct * 64;

    __shared__ _Float16 As[128][LPAD];
    __shared__ _Float16 Bs[64][LPAD];

    int tid = threadIdx.x, lane = tid & 31, w = tid >> 5;
    int wm = w & 3, wn = w >> 2;
    int mrow = lane & 15, grp = lane >> 4;

    v8f acc[2][2] = {};
    const _Float16* ab = ap + (size_t)b * NPIX * C_DIM;

    unsigned ldsA = (unsigned)(size_t)&As[0][0];
    unsigned ldsB = (unsigned)(size_t)&Bs[0][0];

    for (int k0 = 0; k0 < C_DIM; k0 += 32) {
#if USE_TDM
        if (w == 0) {
            tdm_load_2d(ldsA, ab + (size_t)n0 * C_DIM + k0, C_DIM, 32, 128);
            tdm_load_2d(ldsB, Wp + (size_t)c0 * C_DIM + k0, C_DIM, 32, 64);
            __builtin_amdgcn_s_wait_tensorcnt(0);
        }
#else
        #pragma unroll
        for (int j = 0; j < 2; ++j) {
            int u = tid + j * 256;
            int m = u >> 2, kv = (u & 3) * 8;
            *(v8h*)&As[m][kv] =
                *(const v8h*)&ab[(size_t)(n0 + m) * C_DIM + k0 + kv];
        }
        {
            int nc = tid >> 2, kv = (tid & 3) * 8;
            *(v8h*)&Bs[nc][kv] =
                *(const v8h*)&Wp[(size_t)(c0 + nc) * C_DIM + k0 + kv];
        }
#endif
        __syncthreads();

        v16h af[2], bf[2];
        #pragma unroll
        for (int t = 0; t < 2; ++t) {
            int mb = wm * 32 + t * 16 + mrow;
            int nb = wn * 32 + t * 16 + mrow;
            #pragma unroll
            for (int e = 0; e < 16; ++e) {
                int kk = frag_k(e, grp);
                af[t][e] = As[mb][kk];
                bf[t][e] = Bs[nb][kk];
            }
        }
        #pragma unroll
        for (int tm = 0; tm < 2; ++tm)
            #pragma unroll
            for (int tn = 0; tn < 2; ++tn)
                acc[tm][tn] = __builtin_amdgcn_wmma_f32_16x16x32_f16(
                    false, af[tm], false, bf[tn], (short)0, acc[tm][tn], false, false);
        __syncthreads();
    }

    #pragma unroll
    for (int tm = 0; tm < 2; ++tm)
        #pragma unroll
        for (int tn = 0; tn < 2; ++tn)
            #pragma unroll
            for (int r = 0; r < 8; ++r) {
                int n  = n0 + wm * 32 + tm * 16 + r + 8 * grp;
                int co = c0 + wn * 32 + tn * 16 + (lane & 15);
                out[((size_t)b * C_DIM + co) * NPIX + n] = acc[tm][tn][r] + bproj[co];
            }
}

// ===========================================================================
extern "C" void kernel_launch(void* const* d_in, const int* in_sizes, int n_in,
                              void* d_out, int out_size, void* d_ws, size_t ws_size,
                              hipStream_t stream) {
    (void)in_sizes; (void)n_in; (void)out_size; (void)ws_size;
    const float* x           = (const float*)d_in[0];
    const float* Wq          = (const float*)d_in[1];
    const float* Wkv         = (const float*)d_in[2];
    const float* Wproj       = (const float*)d_in[3];
    const float* bproj       = (const float*)d_in[4];
    const float* dwc_w       = (const float*)d_in[5];
    const float* dwc_b       = (const float*)d_in[6];
    const float* scale_param = (const float*)d_in[7];
    float* out = (float*)d_out;

    char* ws = (char*)d_ws;
    size_t off = 0;
    auto alloc = [&](size_t bytes) { size_t r = off; off = (off + bytes + 255) & ~(size_t)255; return r; };
    _Float16* WqkvH  = (_Float16*)(ws + alloc((size_t)1536 * 512 * 2));
    _Float16* WprojH = (_Float16*)(ws + alloc((size_t)512 * 512 * 2));
    float* inv_scale = (float*)(ws + alloc(C_DIM * 4));
    float* ksum      = (float*)(ws + alloc((size_t)BDIM * C_DIM * 4));
    float* kvm       = (float*)(ws + alloc((size_t)BDIM * NH * HD * HD * 4));
    _Float16* kvmH   = (_Float16*)(ws + alloc((size_t)BDIM * NH * HD * HD * 2));
    float* zfac      = (float*)(ws + alloc((size_t)BDIM * NPIX * NH * 4));
    float* qf        = (float*)(ws + alloc((size_t)BDIM * NPIX * C_DIM * 4));  // later: x_attn
    float* kf        = (float*)(ws + alloc((size_t)BDIM * NPIX * C_DIM * 4));  // later: dwc out
    float* vf        = (float*)(ws + alloc((size_t)BDIM * NPIX * C_DIM * 4));
    _Float16* qpH    = (_Float16*)(ws + alloc((size_t)BDIM * NPIX * C_DIM * 2));
    _Float16* kpH    = (_Float16*)(ws + alloc((size_t)BDIM * NPIX * C_DIM * 2));
    _Float16* apH    = (_Float16*)(ws + alloc((size_t)BDIM * NPIX * C_DIM * 2));
    float* xat   = qf;
    float* dwcO  = kf;

    k_prep<<<512, 256, 0, stream>>>(Wq, Wkv, Wproj, scale_param,
                                    WqkvH, WprojH, inv_scale, ksum, kvm);
    k_qkv_gemm<<<BDIM * 32 * 24, 256, 0, stream>>>(x, WqkvH, qf, kf, vf);
    k_transform<<<BDIM * 128, 256, 0, stream>>>(qf, inv_scale, qpH, ksum, 0);
    k_transform<<<BDIM * 128, 256, 0, stream>>>(kf, inv_scale, kpH, ksum, 1);
    k_zfac<<<(BDIM * NPIX * NH + 255) / 256, 256, 0, stream>>>(qpH, ksum, zfac);
    k_kvm_gemm<<<BDIM * NH * 4, 256, 0, stream>>>(kpH, vf, kvm);
    k_kvm_cvt<<<(BDIM * NH * HD * HD + 255) / 256, 256, 0, stream>>>(kvm, kvmH);
    k_attn_out<<<BDIM * NH * 32, 256, 0, stream>>>(qpH, kvmH, zfac, xat);
    k_dwc<<<BDIM * C_DIM * 16, 256, 0, stream>>>(vf, dwc_w, dwc_b, dwcO);
    k_add<<<BDIM * 128 * 16, 256, 0, stream>>>(xat, dwcO, apH);
    k_proj_gemm<<<BDIM * 32 * 8, 256, 0, stream>>>(apH, WprojH, bproj, out);
}
